// InteractionModule_81037442941005
// MI455X (gfx1250) — matrix operand order repaired and checked by reading, not compiled, not run on hardware
//
#include <hip/hip_runtime.h>
#include <hip/hip_bf16.h>
#include <math.h>

typedef __attribute__((ext_vector_type(16))) _Float16 v16h;
typedef __attribute__((ext_vector_type(8)))  _Float16 v8h;
typedef __attribute__((ext_vector_type(8)))  float    v8f;
typedef __attribute__((ext_vector_type(4)))  int      v4i;
typedef _Float16 h16_t;

#define N_NODES 4096
#define E_EDGES 40960
#define R_MAXF  5.0f
#define ENV_NORMF 5.0f
#define ISQRTDF 5.0f

// ---------------------------------------------------------------------------
// Async global->LDS copy (gfx1250). Builtin takes v4i* in AS1/AS3.
// ---------------------------------------------------------------------------
#if __has_builtin(__builtin_amdgcn_global_load_async_to_lds_b128) && \
    __has_builtin(__builtin_amdgcn_s_wait_asynccnt)
#define HAVE_ASYNC_LDS 1
typedef __attribute__((address_space(1))) v4i* gv4_ptr;
typedef __attribute__((address_space(3))) v4i* lv4_ptr;
__device__ __forceinline__ void async_g2l_16B(const void* g, void* l) {
  __builtin_amdgcn_global_load_async_to_lds_b128((gv4_ptr)(void*)g, (lv4_ptr)l,
                                                 0, 0);
}
__device__ __forceinline__ void async_wait0() {
  __builtin_amdgcn_s_wait_asynccnt(0);
}
#else
#define HAVE_ASYNC_LDS 0
#endif

// ---------------------------------------------------------------------------
// WMMA helpers (gfx1250, wave32). Layouts used here (contiguous K order, not the ISA table order):
//  A (16x32 f16): row=lane&15, K base=(lane>>4)*16, 16 consecutive K
//  B (32x16 f16): col=lane&15, K base=(lane>>4)*16, 16 consecutive K
//  C (16x16 f32): col=lane&15, rows rb..rb+7, rb=(lane>>4)*8
// B matrices are PRE-PACKED: lane L's 16 halves contiguous at
// ((ct*nkb+kb)*32 + L)*16.
// ---------------------------------------------------------------------------
__device__ __forceinline__ v8f wmma32(v16h a, v16h b, v8f c) {
  return __builtin_amdgcn_wmma_f32_16x16x32_f16(false, a, false, b, (short)0, c,
                                                false, false);
}

__device__ __forceinline__ v16h loadA(const h16_t* A, int ld, int lane) {
  int row = lane & 15;
  int kl  = (lane >> 4) << 4;
  const h16_t* p = A + row * ld + kl;
  v8h lo = *(const v8h*)p;         // ds_load_b128
  v8h hi = *(const v8h*)(p + 8);   // ds_load_b128
  return __builtin_shufflevector(lo, hi, 0, 1, 2, 3, 4, 5, 6, 7,
                                 8, 9, 10, 11, 12, 13, 14, 15);
}

template <int KB>
__device__ __forceinline__ void loadAfrags(v16h* f, const h16_t* A, int ld, int lane) {
#pragma unroll
  for (int k = 0; k < KB; ++k) f[k] = loadA(A + (k << 5), ld, lane);
}

__device__ __forceinline__ v16h loadBfrag(const h16_t* Bp, int k, int lane) {
  return *(const v16h*)(Bp + (k << 9) + lane * 16);
}

template <int KB>
__device__ __forceinline__ v8f gemmp(const v16h* af, const h16_t* Bp, int lane) {
  v8f acc = {};
#pragma unroll
  for (int k = 0; k < KB; ++k) acc = wmma32(af[k], loadBfrag(Bp, k, lane), acc);
  return acc;
}

__device__ __forceinline__ float rcp_fast(float x) { return __builtin_amdgcn_rcpf(x); }
__device__ __forceinline__ float sigmoidf_(float x) { return rcp_fast(1.0f + __expf(-x)); }
__device__ __forceinline__ float siluf_(float x)    { return x * sigmoidf_(x); }

// ---------------------------------------------------------------------------
// Utility kernels
// ---------------------------------------------------------------------------
__global__ void k_zero(float* p, int n) {
  int i = blockIdx.x * blockDim.x + threadIdx.x;
  if (i < n) p[i] = 0.0f;
}

// fp32 (srcRows x cols) -> f16 packed in WMMA-B fragment order, K padded.
__global__ void k_pack(const float* __restrict__ src, h16_t* __restrict__ dst,
                       int srcRows, int cols, int Kpad) {
  int o = blockIdx.x * blockDim.x + threadIdx.x;
  int nkb = Kpad >> 5;
  int total = (cols >> 4) * nkb * 512;
  if (o >= total) return;
  int i  = o & 15;
  int L  = (o >> 4) & 31;
  int kb = (o >> 9) % nkb;
  int ct = o / (nkb << 9);
  int k   = (kb << 5) + ((L >> 4) << 4) + i;
  int col = (ct << 4) + (L & 15);
  dst[o] = (k < srcRows) ? (h16_t)src[(size_t)k * cols + col] : (h16_t)0.0f;
}

// ---------------------------------------------------------------------------
// latent = silu([na[center] | na[neigh] | radial] @ W_lat0) * cutoff
// one wave = 16 edges; writes f32 latent + f16 mirror
// ---------------------------------------------------------------------------
__global__ void __launch_bounds__(32)
k_latent(const float* __restrict__ node_attrs,
         const float* __restrict__ edge_radial,
         const float* __restrict__ edge_lengths,
         const int*   __restrict__ edge_index,
         const h16_t* __restrict__ Wlat0p,  // packed, Kpad=64 (nkb=2), 8 tiles
         float* __restrict__ latent, h16_t* __restrict__ lath,
         float* __restrict__ cutoff) {
  __shared__ __align__(32) h16_t A[16 * 64];
  __shared__ float cutb[16];
  __shared__ int   cidx[16], nidx[16];
  int lane = threadIdx.x;
  int e0 = blockIdx.x * 16;
  if (lane < 16) {
    int e = e0 + lane;
    cidx[lane] = edge_index[e];
    nidx[lane] = edge_index[E_EDGES + e];
    float x = edge_lengths[e] * (1.0f / R_MAXF);
    float x2 = x * x, x3 = x2 * x, x6 = x3 * x3;
    float c = 1.0f - 28.0f * x6 + 48.0f * x6 * x - 21.0f * x6 * x2;
    c = (x < 1.0f) ? c : 0.0f;
    cutb[lane] = c;
    cutoff[e] = c;
  }
  __syncthreads();
  for (int idx = lane; idx < 16 * 64; idx += 32) {
    int r = idx >> 6, c = idx & 63;
    float v;
    if (c < 16)      v = node_attrs[cidx[r] * 16 + c];
    else if (c < 32) v = node_attrs[nidx[r] * 16 + (c - 16)];
    else if (c < 40) v = edge_radial[(size_t)(e0 + r) * 8 + (c - 32)];
    else             v = 0.0f;
    A[idx] = (h16_t)v;
  }
  __syncthreads();
  v16h af[2];
  loadAfrags<2>(af, A, 64, lane);
  int l15 = lane & 15, rb = (lane >> 4) << 3;
  for (int ct = 0; ct < 8; ++ct) {
    v8f acc = gemmp<2>(af, Wlat0p + ct * 1024, lane);
    int col = ct * 16 + l15;
#pragma unroll
    for (int i = 0; i < 8; ++i) {
      float v = siluf_(acc[i]) * cutb[rb + i];
      size_t o = (size_t)(e0 + rb + i) * 128 + col;
      latent[o] = v;
      lath[o] = (h16_t)v;
    }
  }
}

// ---------------------------------------------------------------------------
// Pass A (2 edge-tiles per wave): wts GEMM, emb/gate split, feats (l==0),
// attention logits via WMMA q/k tiles (B fragments shared across both tiles),
// segment-sum env via atomics.
// ---------------------------------------------------------------------------
__global__ void __launch_bounds__(32)
k_passA(int l,
        const float* __restrict__ node_attrs,
        const float* __restrict__ edge_angular,
        const int*   __restrict__ edge_index,
        const h16_t* __restrict__ lath,
        const float* __restrict__ cutoff,
        const h16_t* __restrict__ Wenv,  // packed, Kpad=128 (nkb=4)
        const h16_t* __restrict__ Wq,    // packed, Kpad=32 (nkb=1), 128 tiles
        const h16_t* __restrict__ Wk,    // packed, Kpad=64 (nkb=2), 128 tiles
        float* __restrict__ feat_s,      // E x 64      (written when l==0)
        float* __restrict__ feat_v,      // E x 64 x 3  (written when l==0)
        float* __restrict__ env_s,       // N x 64      (atomic accumulate)
        float* __restrict__ env_v) {     // N x 64 x 3  (atomic accumulate)
  __shared__ __align__(32) h16_t latA[32 * 128];  // tile u at u*2048
  __shared__ __align__(32) h16_t xA[32 * 64];     // tile u at u*1024
  __shared__ __align__(32) h16_t qA[32 * 32];     // tile u at u*512
  __shared__ float embs_t[32 * 64];
  __shared__ float wenv1_t[32 * 64];
  __shared__ float gate_t[32 * 64];
  __shared__ float g_t[32 * 64];
  __shared__ float sh0[32];
  __shared__ float sh1[32 * 3];
  __shared__ float cutb[32];
  __shared__ int   cidx[32];
  int lane = threadIdx.x;
  int e0 = blockIdx.x * 32;
  {
    int e = e0 + lane;
    cidx[lane] = edge_index[e];
    sh0[lane] = edge_angular[(size_t)e * 4 + 0];
    sh1[lane * 3 + 0] = edge_angular[(size_t)e * 4 + 1];
    sh1[lane * 3 + 1] = edge_angular[(size_t)e * 4 + 2];
    sh1[lane * 3 + 2] = edge_angular[(size_t)e * 4 + 3];
    cutb[lane] = cutoff[e];
  }
  __syncthreads();
  // latA: contiguous f16 copy of 32 latent rows (4096 halves)
#if HAVE_ASYNC_LDS
  for (int it = 0; it < 16; ++it) {
    int off = (it * 32 + lane) * 8;  // halves (16B granules)
    async_g2l_16B(lath + (size_t)e0 * 128 + off, latA + off);
  }
#else
  for (int it = lane; it < 512; it += 32)
    *(v8h*)(latA + it * 8) = *(const v8h*)(lath + (size_t)e0 * 128 + it * 8);
#endif
  for (int idx = lane; idx < 32 * 32; idx += 32) {
    int r = idx >> 5, c = idx & 31;
    qA[idx] = (h16_t)((c < 16) ? node_attrs[cidx[r] * 16 + c] : 0.0f);
  }
#if HAVE_ASYNC_LDS
  async_wait0();
#endif
  __syncthreads();

  const int ncolw = (l == 0) ? 320 : 192;
  const int ebase = (l == 0) ? 128 : 0;
  const int gbase = (l == 0) ? 256 : 128;
  int l15 = lane & 15, rb = (lane >> 4) << 3;

  v16h la0[4], la1[4];
  loadAfrags<4>(la0, latA, 128, lane);
  loadAfrags<4>(la1, latA + 16 * 128, 128, lane);
  for (int ct = 0; ct < ncolw / 16; ++ct) {
    v8f acc0 = {}, acc1 = {};
#pragma unroll
    for (int k = 0; k < 4; ++k) {
      v16h b = loadBfrag(Wenv + ct * 2048, k, lane);
      acc0 = wmma32(la0[k], b, acc0);
      acc1 = wmma32(la1[k], b, acc1);
    }
    int col0 = ct * 16;
    int col = col0 + l15;
    auto scatter = [&](const v8f& acc, int u) {
      int rB = u * 16 + rb;      // row base in block-local buffers
      int eB = e0 + rB;          // edge base
      if (l == 0 && col0 < 128) {            // w_feat -> feat_s / feat_v
        int m = col >> 1, ii = col & 1;
        if (ii == 0) {
#pragma unroll
          for (int i = 0; i < 8; ++i)
            feat_s[(size_t)(eB + i) * 64 + m] = acc[i] * sh0[rB + i];
        } else {
#pragma unroll
          for (int i = 0; i < 8; ++i) {
            size_t o = ((size_t)(eB + i) * 64 + m) * 3;
            feat_v[o + 0] = acc[i] * sh1[(rB + i) * 3 + 0];
            feat_v[o + 1] = acc[i] * sh1[(rB + i) * 3 + 1];
            feat_v[o + 2] = acc[i] * sh1[(rB + i) * 3 + 2];
          }
        }
      } else if (col0 < ebase + 128) {       // w_env -> emb_s / w_env1
        int cc = col - ebase, m = cc >> 1, ii = cc & 1;
        if (ii == 0) {
#pragma unroll
          for (int i = 0; i < 8; ++i) embs_t[(rB + i) * 64 + m] = acc[i] * sh0[rB + i];
        } else {
#pragma unroll
          for (int i = 0; i < 8; ++i) wenv1_t[(rB + i) * 64 + m] = acc[i];
        }
      } else {                               // w_gate
        int m = col - gbase;
#pragma unroll
        for (int i = 0; i < 8; ++i) gate_t[(rB + i) * 64 + m] = acc[i];
      }
    };
    scatter(acc0, 0);
    scatter(acc1, 1);
  }
  __syncthreads();
  for (int idx = lane; idx < 32 * 64; idx += 32)
    xA[idx] = (h16_t)(embs_t[idx] * gate_t[idx]);
  __syncthreads();

  // attention: logits[e,m] = sum_d q[e,m,d]*k[e,m,d]; B frags shared by tiles
  v16h aq0, aq1, ax0[2], ax1[2];
  loadAfrags<1>(&aq0, qA, 32, lane);
  loadAfrags<1>(&aq1, qA + 512, 32, lane);
  loadAfrags<2>(ax0, xA, 64, lane);
  loadAfrags<2>(ax1, xA + 1024, 64, lane);
  for (int m = 0; m < 64; ++m) {
    float p0[8], p1[8];
#pragma unroll
    for (int i = 0; i < 8; ++i) { p0[i] = 0.0f; p1[i] = 0.0f; }
#pragma unroll
    for (int sub = 0; sub < 2; ++sub) {
      int ct = 2 * m + sub;
      if (ct + 2 < 128) {  // stream next fragment blocks through L2
        __builtin_prefetch(Wk + (size_t)(ct + 2) * 1024 + lane * 16, 0, 1);
        __builtin_prefetch(Wq + (size_t)(ct + 2) * 512 + lane * 16, 0, 1);
      }
      v8f z = {};
      v16h bq = *(const v16h*)(Wq + (size_t)ct * 512 + lane * 16);
      v8f q0 = wmma32(aq0, bq, z);
      v8f q1 = wmma32(aq1, bq, z);
      v8f k0 = z, k1 = z;
#pragma unroll
      for (int kk = 0; kk < 2; ++kk) {
        v16h bk = loadBfrag(Wk + (size_t)ct * 1024, kk, lane);
        k0 = wmma32(ax0[kk], bk, k0);
        k1 = wmma32(ax1[kk], bk, k1);
      }
#pragma unroll
      for (int i = 0; i < 8; ++i) {
        p0[i] += k0[i] * q0[i];
        p1[i] += k1[i] * q1[i];
      }
    }
#pragma unroll
    for (int mask = 1; mask < 16; mask <<= 1)
#pragma unroll
      for (int i = 0; i < 8; ++i) {
        p0[i] += __shfl_xor(p0[i], mask, 32);
        p1[i] += __shfl_xor(p1[i], mask, 32);
      }
    if (l15 == 0) {
#pragma unroll
      for (int i = 0; i < 8; ++i) {
        g_t[(rb + i) * 64 + m]      = sigmoidf_(p0[i] * (1.0f / ISQRTDF)) * cutb[rb + i];
        g_t[(16 + rb + i) * 64 + m] = sigmoidf_(p1[i] * (1.0f / ISQRTDF)) * cutb[16 + rb + i];
      }
    }
  }
  __syncthreads();

  for (int idx = lane; idx < 32 * 64; idx += 32) {
    int r = idx >> 6, m = idx & 63;
    float g = g_t[idx];
    int n = cidx[r];
    atomicAdd(&env_s[(size_t)n * 64 + m], embs_t[idx] * g);
    float wv = wenv1_t[idx] * g;
    size_t o = ((size_t)n * 64 + m) * 3;
    atomicAdd(&env_v[o + 0], wv * sh1[r * 3 + 0]);
    atomicAdd(&env_v[o + 1], wv * sh1[r * 3 + 1]);
    atomicAdd(&env_v[o + 2], wv * sh1[r * 3 + 2]);
  }
}

// ---------------------------------------------------------------------------
// Pass N (per 16 nodes): es = (env_s/5 @ Wenvlin0) * (na @ Wprod)
//                        ev[:,k,c] = (env_v[:,:,c]/5 @ Wenvlin1) * gate
// ---------------------------------------------------------------------------
__global__ void __launch_bounds__(32)
k_passN(const float* __restrict__ node_attrs,
        const float* __restrict__ env_s,
        const float* __restrict__ env_v,
        const h16_t* __restrict__ Wprod,     // packed, Kpad=32 (nkb=1), 4 tiles
        const h16_t* __restrict__ Wenvlin0,  // packed, Kpad=64 (nkb=2), 4 tiles
        const h16_t* __restrict__ Wenvlin1,  // packed, Kpad=64 (nkb=2), 4 tiles
        float* __restrict__ es, float* __restrict__ ev) {
  __shared__ __align__(32) h16_t naA[16 * 32];
  __shared__ __align__(32) h16_t svA[16 * 64];
  __shared__ float gate_t[16 * 64];
  int lane = threadIdx.x;
  int n0 = blockIdx.x * 16;
  for (int idx = lane; idx < 16 * 32; idx += 32) {
    int r = idx >> 5, c = idx & 31;
    naA[idx] = (h16_t)((c < 16) ? node_attrs[(n0 + r) * 16 + c] : 0.0f);
  }
  for (int idx = lane; idx < 16 * 64; idx += 32) {
    int r = idx >> 6, c = idx & 63;
    svA[idx] = (h16_t)(env_s[(size_t)(n0 + r) * 64 + c] * (1.0f / ENV_NORMF));
  }
  __syncthreads();
  int l15 = lane & 15, rb = (lane >> 4) << 3;
  v16h ana;
  loadAfrags<1>(&ana, naA, 32, lane);
  v16h asv[2];
  loadAfrags<2>(asv, svA, 64, lane);
  for (int ct = 0; ct < 4; ++ct) {
    v8f gacc = gemmp<1>(&ana, Wprod + ct * 512, lane);
    v8f sacc = gemmp<2>(asv, Wenvlin0 + ct * 1024, lane);
    int col = ct * 16 + l15;
#pragma unroll
    for (int i = 0; i < 8; ++i) {
      gate_t[(rb + i) * 64 + col] = gacc[i];
      es[(size_t)(n0 + rb + i) * 64 + col] = sacc[i] * gacc[i];
    }
  }
  __syncthreads();
  for (int c = 0; c < 3; ++c) {
    for (int idx = lane; idx < 16 * 64; idx += 32) {
      int r = idx >> 6, cc = idx & 63;
      svA[idx] = (h16_t)(env_v[((size_t)(n0 + r) * 64 + cc) * 3 + c] * (1.0f / ENV_NORMF));
    }
    __syncthreads();
    loadAfrags<2>(asv, svA, 64, lane);
    for (int ct = 0; ct < 4; ++ct) {
      v8f vacc = gemmp<2>(asv, Wenvlin1 + ct * 1024, lane);
      int k = ct * 16 + l15;
#pragma unroll
      for (int i = 0; i < 8; ++i)
        ev[((size_t)(n0 + rb + i) * 64 + k) * 3 + c] = vacc[i] * gate_t[(rb + i) * 64 + k];
    }
    __syncthreads();
  }
}

// ---------------------------------------------------------------------------
// Pass B: product basis + W_lin_s/W_lin_v + latent update (l==0) / W_final
// W_lin_v B-fragments loaded once per column tile, shared by all 3 components.
// ---------------------------------------------------------------------------
__global__ void __launch_bounds__(32)
k_passB(int l,
        const int*   __restrict__ edge_index,
        const float* __restrict__ cutoff,
        const float* __restrict__ es,
        const float* __restrict__ ev,
        const h16_t* __restrict__ Wlins,  // packed, Kpad=128 (nkb=4), 4 tiles
        const h16_t* __restrict__ Wlinv,  // packed, Kpad=192 (nkb=6), 4 tiles
        const h16_t* __restrict__ Wlat1,  // packed, Kpad=256 (nkb=8), 8 tiles
        const h16_t* __restrict__ Wfin,   // packed, Kpad=256 (nkb=8), 4 tiles
        const float* __restrict__ alpha,
        float* __restrict__ latent, h16_t* __restrict__ lath,
        float* __restrict__ feat_s,
        float* __restrict__ feat_v,
        float* __restrict__ out) {
  __shared__ __align__(32) h16_t latinA[16 * 256];
  __shared__ __align__(32) h16_t scalA[16 * 128];
  __shared__ __align__(32) h16_t vecA[3 * 16 * 192];  // component c at c*3072
  __shared__ float fs_t[16 * 64];
  __shared__ float es_t[16 * 64];
  __shared__ h16_t fv_t[16 * 192];
  __shared__ h16_t ev_t[16 * 192];
  __shared__ float cutb[16];
  __shared__ int   cidx[16];
  int lane = threadIdx.x;
  int e0 = blockIdx.x * 16;
  if (lane < 16) {
    int e = e0 + lane;
    cidx[lane] = edge_index[e];
    cutb[lane] = cutoff[e];
  }
  __syncthreads();
  // latinA cols 0..127 <- f16 latent rows (16B copies)
  for (int it = lane; it < 256; it += 32) {
    int r = it >> 4, g = it & 15;
    *(v8h*)(latinA + r * 256 + g * 8) = *(const v8h*)(lath + (size_t)(e0 + r) * 128 + g * 8);
  }
  for (int idx = lane; idx < 16 * 64; idx += 32) {
    int r = idx >> 6, m = idx & 63;
    fs_t[idx] = feat_s[(size_t)(e0 + r) * 64 + m];
    es_t[idx] = es[(size_t)cidx[r] * 64 + m];
  }
  for (int idx = lane; idx < 16 * 192; idx += 32) {
    int r = idx / 192, q = idx - r * 192;  // q = m*3 + c
    fv_t[idx] = (h16_t)feat_v[(size_t)(e0 + r) * 192 + q];
    ev_t[idx] = (h16_t)ev[(size_t)cidx[r] * 192 + q];
  }
  __syncthreads();
  for (int idx = lane; idx < 16 * 64; idx += 32) {
    int r = idx >> 6, m = idx & 63;
    float fs = fs_t[idx], esv = es_t[idx];
    float fv[3], evv[3];
#pragma unroll
    for (int c = 0; c < 3; ++c) {
      fv[c]  = (float)fv_t[r * 192 + m * 3 + c];
      evv[c] = (float)ev_t[r * 192 + m * 3 + c];
    }
    float s1 = fs * esv;
    float s2 = (fv[0] * evv[0] + fv[1] * evv[1] + fv[2] * evv[2]) * 0.5773502691896258f;
    scalA[r * 128 + m]        = (h16_t)s1;
    scalA[r * 128 + 64 + m]   = (h16_t)s2;
    latinA[r * 256 + 128 + 2 * m + 0] = (h16_t)s1;   // sc_flat: (m, p) interleave
    latinA[r * 256 + 128 + 2 * m + 1] = (h16_t)s2;
#pragma unroll
    for (int c = 0; c < 3; ++c) {
      int c1 = (c + 1 < 3) ? c + 1 : c - 2;
      int c2 = (c + 2 < 3) ? c + 2 : c - 1;
      h16_t* vA = vecA + c * 3072;
      vA[r * 192 + m]       = (h16_t)(fs * evv[c]);                          // v1
      vA[r * 192 + 64 + m]  = (h16_t)(fv[c] * esv);                          // v2
      vA[r * 192 + 128 + m] = (h16_t)((fv[c1] * evv[c2] - fv[c2] * evv[c1])
                                      * 0.7071067811865476f);                // v3
    }
  }
  __syncthreads();
  int l15 = lane & 15, rb = (lane >> 4) << 3;
  if (l == 0) {
    v16h as[4];
    loadAfrags<4>(as, scalA, 128, lane);
    for (int ct = 0; ct < 4; ++ct) {
      v8f acc = gemmp<4>(as, Wlins + ct * 2048, lane);
      int k = ct * 16 + l15;
#pragma unroll
      for (int i = 0; i < 8; ++i)
        feat_s[(size_t)(e0 + rb + i) * 64 + k] = acc[i] * 0.08838834764831845f; // 1/sqrt(2M)
    }
  }
  for (int ct = 0; ct < 4; ++ct) {
    v16h bf[6];
#pragma unroll
    for (int k = 0; k < 6; ++k) bf[k] = loadBfrag(Wlinv + ct * 3072, k, lane);
#pragma unroll
    for (int c = 0; c < 3; ++c) {
      v8f acc = {};
#pragma unroll
      for (int k = 0; k < 6; ++k)
        acc = wmma32(loadA(vecA + c * 3072 + (k << 5), 192, lane), bf[k], acc);
      int kc = ct * 16 + l15;
#pragma unroll
      for (int i = 0; i < 8; ++i) {
        float v = acc[i] * 0.07216878364870323f;  // 1/sqrt(3M)
        if (l == 0) feat_v[((size_t)(e0 + rb + i) * 64 + kc) * 3 + c] = v;
        else        out[(size_t)(e0 + rb + i) * 256 + 64 + kc * 3 + c] = v;
      }
    }
  }
  v16h alat[8];
  loadAfrags<8>(alat, latinA, 256, lane);
  if (l == 0) {
    float al = alpha[0];
    float co = rsqrtf(1.0f + al * al);
    for (int ct = 0; ct < 8; ++ct) {
      v8f acc = gemmp<8>(alat, Wlat1 + ct * 4096, lane);
      int col = ct * 16 + l15;
#pragma unroll
      for (int i = 0; i < 8; ++i) {
        size_t o = (size_t)(e0 + rb + i) * 128 + col;
        float nl = siluf_(acc[i]) * cutb[rb + i];
        float nv = co * latent[o] + al * co * nl;
        latent[o] = nv;
        lath[o] = (h16_t)nv;
      }
    }
  } else {
    for (int ct = 0; ct < 4; ++ct) {
      v8f acc = gemmp<8>(alat, Wfin + ct * 4096, lane);
      int col = ct * 16 + l15;
#pragma unroll
      for (int i = 0; i < 8; ++i)
        out[(size_t)(e0 + rb + i) * 256 + col] = acc[i];
    }
  }
}

// ---------------------------------------------------------------------------
extern "C" void kernel_launch(void* const* d_in, const int* in_sizes, int n_in,
                              void* d_out, int out_size, void* d_ws, size_t ws_size,
                              hipStream_t stream) {
  (void)in_sizes; (void)n_in; (void)out_size; (void)ws_size;
  const float* node_attrs   = (const float*)d_in[0];
  const float* edge_radial  = (const float*)d_in[1];
  const float* edge_angular = (const float*)d_in[2];
  const float* edge_lengths = (const float*)d_in[3];
  const int*   edge_index   = (const int*)d_in[4];
  const float* W_lat0   = (const float*)d_in[5];
  const float* W_lat1   = (const float*)d_in[6];
  const float* W_final  = (const float*)d_in[7];
  const float* W_env0   = (const float*)d_in[8];
  const float* W_env1   = (const float*)d_in[9];
  const float* W_q      = (const float*)d_in[10];
  const float* W_k      = (const float*)d_in[11];
  const float* W_envlin = (const float*)d_in[12];
  const float* W_prod   = (const float*)d_in[13];
  const float* W_lin_s  = (const float*)d_in[14];
  const float* W_lin_v  = (const float*)d_in[15];
  const float* alpha    = (const float*)d_in[16];
  float* out = (float*)d_out;

  // ---- workspace layout ----
  h16_t* hbase = (h16_t*)d_ws;
  size_t ho = 0;
  auto allocH = [&](size_t n) { h16_t* p = hbase + ho; ho += n; return p; };
  h16_t* wlat0h   = allocH(64 * 128);                 // Kpad=64
  h16_t* wlat1h   = allocH(256 * 128);                // Kpad=256
  h16_t* wfinh    = allocH(256 * 64);                 // Kpad=256
  h16_t* wenv0h   = allocH(128 * 320);                // Kpad=128
  h16_t* wenv1h   = allocH(128 * 192);                // Kpad=128
  h16_t* wqh      = allocH((size_t)2 * 32 * 2048);    // Kpad=32
  h16_t* wkh      = allocH((size_t)2 * 64 * 2048);    // Kpad=64
  h16_t* wenvlinh = allocH((size_t)2 * 2 * 64 * 64);  // Kpad=64
  h16_t* wprodh   = allocH((size_t)2 * 32 * 64);      // Kpad=32
  h16_t* wlinsh   = allocH((size_t)2 * 128 * 64);     // Kpad=128
  h16_t* wlinvh   = allocH((size_t)2 * 192 * 64);     // Kpad=192
  h16_t* lath     = allocH((size_t)E_EDGES * 128);    // f16 latent mirror
  ho = (ho + 15) & ~(size_t)15;
  float* fbase = (float*)(hbase + ho);
  size_t fo = 0;
  auto allocF = [&](size_t n) { float* p = fbase + fo; fo += n; return p; };
  float* latent = allocF((size_t)E_EDGES * 128);
  float* cutoff = allocF(E_EDGES);
  float* feat_s = allocF((size_t)E_EDGES * 64);
  float* feat_v = allocF((size_t)E_EDGES * 192);
  float* env_s  = allocF((size_t)N_NODES * 64);
  float* env_v  = allocF((size_t)N_NODES * 192);
  float* es     = allocF((size_t)N_NODES * 64);
  float* ev     = allocF((size_t)N_NODES * 192);

  // ---- weight conversion: fp32 -> f16 fragment-packed, K padded ----
  auto pack = [&](const float* s, h16_t* d, int sr, int c, int kpad) {
    int tot = (c >> 4) * (kpad >> 5) * 512;
    k_pack<<<(tot + 255) / 256, 256, 0, stream>>>(s, d, sr, c, kpad);
  };
  pack(W_lat0, wlat0h, 40, 128, 64);
  pack(W_lat1, wlat1h, 256, 128, 256);
  pack(W_final, wfinh, 256, 64, 256);
  pack(W_env0, wenv0h, 128, 320, 128);
  pack(W_env1, wenv1h, 128, 192, 128);
  for (int l = 0; l < 2; ++l) {
    pack(W_q + (size_t)l * 16 * 2048, wqh + (size_t)l * 32 * 2048, 16, 2048, 32);
    pack(W_k + (size_t)l * 64 * 2048, wkh + (size_t)l * 64 * 2048, 64, 2048, 64);
    pack(W_prod + (size_t)l * 16 * 64, wprodh + (size_t)l * 32 * 64, 16, 64, 32);
    pack(W_envlin + (size_t)(2 * l) * 64 * 64, wenvlinh + (size_t)(2 * l) * 64 * 64, 64, 64, 64);
    pack(W_envlin + (size_t)(2 * l + 1) * 64 * 64, wenvlinh + (size_t)(2 * l + 1) * 64 * 64, 64, 64, 64);
    pack(W_lin_s + (size_t)l * 128 * 64, wlinsh + (size_t)l * 128 * 64, 128, 64, 128);
    pack(W_lin_v + (size_t)l * 192 * 64, wlinvh + (size_t)l * 192 * 64, 192, 64, 192);
  }

  // ---- latent precompute ----
  k_latent<<<E_EDGES / 16, 32, 0, stream>>>(node_attrs, edge_radial, edge_lengths,
                                            edge_index, wlat0h, latent, lath, cutoff);

  // ---- layers ----
  for (int l = 0; l < 2; ++l) {
    int nz1 = N_NODES * 64;
    int nz2 = N_NODES * 192;
    k_zero<<<(nz1 + 255) / 256, 256, 0, stream>>>(env_s, nz1);
    k_zero<<<(nz2 + 255) / 256, 256, 0, stream>>>(env_v, nz2);
    const h16_t* Wenv = (l == 0) ? wenv0h : wenv1h;
    k_passA<<<E_EDGES / 32, 32, 0, stream>>>(l, node_attrs, edge_angular, edge_index,
                                             lath, cutoff, Wenv,
                                             wqh + (size_t)l * 32 * 2048,
                                             wkh + (size_t)l * 64 * 2048,
                                             feat_s, feat_v, env_s, env_v);
    k_passN<<<N_NODES / 16, 32, 0, stream>>>(node_attrs, env_s, env_v,
                                             wprodh + (size_t)l * 32 * 64,
                                             wenvlinh + (size_t)(2 * l) * 64 * 64,
                                             wenvlinh + (size_t)(2 * l + 1) * 64 * 64,
                                             es, ev);
    k_passB<<<E_EDGES / 16, 32, 0, stream>>>(l, edge_index, cutoff, es, ev,
                                             wlinsh + (size_t)l * 128 * 64,
                                             wlinvh + (size_t)l * 192 * 64,
                                             wlat1h, wfinh, alpha,
                                             latent, lath, feat_s, feat_v, out);
  }
}
